// ChannelAttention_18794776887897
// MI455X (gfx1250) — matrix operand order repaired and checked by reading, not compiled, not run on hardware
//
#include <hip/hip_runtime.h>

// ---------------- problem constants ----------------
#define B_    128
#define S_    32
#define EMB_  256
#define IDF_  256
#define C_    1024
#define P_    256          // IH*IW == idf
#define CT_   64           // c-tile width in fused kernel

typedef __attribute__((ext_vector_type(16))) __bf16        bf16x16;
typedef __attribute__((ext_vector_type(2)))  __bf16        bf16x2;
typedef __attribute__((ext_vector_type(8)))  float         f32x8;
typedef __attribute__((ext_vector_type(4)))  float         f32x4;
typedef __attribute__((ext_vector_type(2)))  float         f32x2;
typedef __attribute__((ext_vector_type(4)))  unsigned int  uint4v;
typedef __attribute__((ext_vector_type(2)))  unsigned int  uint2v;
typedef __attribute__((ext_vector_type(4)))  int           int4v;

// padded row strides (ushort units): multiples of 8 -> 16B-aligned b128 rows
#define WE_STRIDE   264   // 256 + 8
#define WCT_STRIDE  264   // 256 + 8
#define KT_STRIDE   40    // 32 + 8
#define WET_STRIDE  40    // 32 + 8
#define ATT_STRIDE  40    // 32 + 8

// packed f32x2 -> bf16x2 in one dword; standard conversion path so the
// backend can select the native packed converter when available
__device__ __forceinline__ unsigned int pack2_bf16(float f0, float f1) {
  f32x2 f = {f0, f1};
  return __builtin_bit_cast(unsigned int, __builtin_convertvector(f, bf16x2));
}

// convert a pair and scatter to two LDS rows (b16 + b16_d16_hi stores)
__device__ __forceinline__ void cvt2_store(unsigned short* d0, unsigned short* d1,
                                           float f0, float f1) {
  const unsigned int u = pack2_bf16(f0, f1);
  *d0 = (unsigned short)u;
  *d1 = (unsigned short)(u >> 16);
}

// fragment = two contiguous 16-byte LDS loads (lowers to ds_load_b128 x2)
__device__ __forceinline__ bf16x16 frag_ld(const unsigned short* lo,
                                           const unsigned short* hi) {
  union { uint4v q[2]; bf16x16 v; } u;
  u.q[0] = *(const uint4v*)lo;
  u.q[1] = *(const uint4v*)hi;
  return u.v;
}

// ------------- async global->LDS (guarded; falls back to b128 copies) -------------
#if defined(__gfx1250__) &&                                          \
    __has_builtin(__builtin_amdgcn_global_load_async_to_lds_b128) && \
    __has_builtin(__builtin_amdgcn_s_wait_asynccnt)
#define HAVE_ASYNC_LDS 1
typedef __attribute__((address_space(1))) int4v ga_int4v;   // global int4
typedef __attribute__((address_space(3))) int4v ls_int4v;   // LDS int4
__device__ __forceinline__ void copy16_async(const void* g, void* l) {
  __builtin_amdgcn_global_load_async_to_lds_b128(
      (ga_int4v*)g, (ls_int4v*)l, 0, 0);
}
__device__ __forceinline__ void async_wait0() {
  __builtin_amdgcn_s_wait_asynccnt(0);
}
#else
#define HAVE_ASYNC_LDS 0
__device__ __forceinline__ void copy16_async(const void* g, void* l) {
  *(uint4v*)l = *(const uint4v*)g;
}
__device__ __forceinline__ void async_wait0() {}
#endif

// =====================================================================
// Kernel 1: we[b] = word_emb[b] @ conv_kernel + bias
//   -> we_out  [B][32][256]  bf16 (row-major, GEMM1 B operand)
//   -> weT_out [B][256][32]  bf16 (transposed, GEMM2 A operand)
// per workgroup: one batch; M=32, N=256, K=256; 8 waves
// =====================================================================
__global__ __launch_bounds__(256) void we_proj_kernel(
    const float* __restrict__ word_emb, const float* __restrict__ ck,
    const float* __restrict__ bias,
    unsigned short* __restrict__ we_out, unsigned short* __restrict__ weT_out)
{
  extern __shared__ char smem1[];
  unsigned short* lds_a  = (unsigned short*)smem1;                        // [32][264]
  unsigned short* lds_kT = (unsigned short*)(smem1 + S_ * WE_STRIDE * 2); // [256][40]

  const int tid  = threadIdx.x;
  const int lane = tid & 31;
  const int wv   = tid >> 5;
  const int half = lane >> 4;
  const int ln   = lane & 15;
  const int b    = blockIdx.x;

  // stage word_emb[b] -> bf16 (contiguous rows: pack pairs, b64 stores)
  const float* asrc = word_emb + (size_t)b * (S_ * EMB_);
  for (int q = tid; q < 2048; q += 256) {
    const int s = q >> 6, x = (q & 63) * 4;
    f32x4 f = *(const f32x4*)(asrc + s * EMB_ + x);
    uint2v u = { pack2_bf16(f[0], f[1]), pack2_bf16(f[2], f[3]) };
    *(uint2v*)(lds_a + s * WE_STRIDE + x) = u;
  }

  f32x8 acc[2][2];
  #pragma unroll
  for (int mt = 0; mt < 2; ++mt)
    #pragma unroll
    for (int q = 0; q < 2; ++q)
      acc[mt][q] = f32x8{0.f,0.f,0.f,0.f,0.f,0.f,0.f,0.f};

  for (int kk = 0; kk < 8; ++kk) {
    const int k0 = kk * 32;
    __syncthreads();
    // stage conv_kernel rows k0..k0+31 transposed: lds_kT[i][kl]
    const float* ksrc = ck + (size_t)k0 * IDF_;
    for (int q = tid; q < 2048; q += 256) {
      const int kl = q >> 6, x = (q & 63) * 4;
      f32x4 f = *(const f32x4*)(ksrc + kl * IDF_ + x);
      cvt2_store(lds_kT + (x    ) * KT_STRIDE + kl,
                 lds_kT + (x + 1) * KT_STRIDE + kl, f[0], f[1]);
      cvt2_store(lds_kT + (x + 2) * KT_STRIDE + kl,
                 lds_kT + (x + 3) * KT_STRIDE + kl, f[2], f[3]);
    }
    __syncthreads();

    bf16x16 afr[2];
    #pragma unroll
    for (int mt = 0; mt < 2; ++mt) {
      const unsigned short* ab = lds_a + (mt * 16 + ln) * WE_STRIDE + k0 + 8 * half;
      afr[mt] = frag_ld(ab, ab + 16);
    }
    #pragma unroll
    for (int q = 0; q < 2; ++q) {
      const int n = (wv + 8 * q) * 16 + ln;
      const unsigned short* bb = lds_kT + n * KT_STRIDE + 8 * half;
      bf16x16 bfr = frag_ld(bb, bb + 16);
      #pragma unroll
      for (int mt = 0; mt < 2; ++mt)
        acc[mt][q] = __builtin_amdgcn_wmma_f32_16x16x32_bf16(
            false, afr[mt], false, bfr, (short)0, acc[mt][q], false, false);
    }
  }

  // epilogue: + bias; pack pairs; weT gets b32 stores (s contiguous), we gets b16
  for (int q = 0; q < 2; ++q) {
    const int n = (wv + 8 * q) * 16 + ln;
    const float bn = bias[n];
    #pragma unroll
    for (int mt = 0; mt < 2; ++mt) {
      const int s0 = mt * 16 + 8 * half;
      unsigned short* wrow = we_out + (size_t)b * (S_ * IDF_) + n;
      unsigned int*   trow = (unsigned int*)(weT_out + ((size_t)b * P_ + n) * S_ + s0);
      #pragma unroll
      for (int i = 0; i < 8; i += 2) {
        const unsigned int u = pack2_bf16(acc[mt][q][i] + bn, acc[mt][q][i + 1] + bn);
        wrow[(size_t)(s0 + i)     * IDF_] = (unsigned short)u;
        wrow[(size_t)(s0 + i + 1) * IDF_] = (unsigned short)(u >> 16);
        trow[i >> 1] = u;
      }
    }
  }
}

// =====================================================================
// Kernel 2 (fused, single pass over wc):
//   logits[c,s] = sum_p wc[b,p,c]*we[s,p]   (WMMA, K=256)
//   attn = softmax_s(logits)
//   out[p,c]   = sum_s we[s,p]*attn[c,s]    (WMMA, K=32)
// grid: B * (C/CT) workgroups of 256 threads (8 waves)
// =====================================================================
__global__ __launch_bounds__(256) void fused_attn_kernel(
    const float* __restrict__ wc,              // [B, 256, 1024] (p, c)
    const unsigned short* __restrict__ we,     // [B, 32, 256] bf16
    const unsigned short* __restrict__ weT,    // [B, 256, 32] bf16
    float* __restrict__ out,                   // [B, 256, 1024]
    float* __restrict__ attn_out)              // [B, 1024, 32]
{
  extern __shared__ char smem[];
  unsigned short* lds_we   = (unsigned short*)(smem);            // [32][264]  16896 B
  unsigned short* lds_weT  = (unsigned short*)(smem + 16896);    // [256][40]  20480 B
  unsigned short* lds_wcT  = (unsigned short*)(smem + 37376);    // [64][264]  33792 B
  float*          lds_lg   = (float*)         (smem + 71168);    // [64][32]    8192 B
  unsigned short* lds_attn = (unsigned short*)(smem + 79360);    // [64][40]    5120 B

  const int tid  = threadIdx.x;
  const int lane = tid & 31;
  const int wv   = tid >> 5;
  const int half = lane >> 4;
  const int ln   = lane & 15;

  const int b  = blockIdx.x >> 4;                 // C_/CT_ = 16 tiles
  const int c0 = (blockIdx.x & 15) * CT_;

  // ---- stage we[b] (bf16 verbatim, padded rows): async b128 ----
  {
    const char* src = (const char*)(we + (size_t)b * (S_ * P_));
    char* dst = (char*)lds_we;
    for (int q = tid; q < 1024; q += 256) {
      const int s = q >> 5, x = (q & 31) * 16;
      copy16_async(src + s * 512 + x, dst + s * (WE_STRIDE * 2) + x);
    }
  }
  // ---- stage weT[b] (bf16 verbatim, padded rows): async b128 ----
  {
    const char* src = (const char*)(weT + (size_t)b * (P_ * S_));
    char* dst = (char*)lds_weT;
    for (int q = tid; q < 1024; q += 256) {
      const int p = q >> 2, x = (q & 3) * 16;
      copy16_async(src + p * 64 + x, dst + p * (WET_STRIDE * 2) + x);
    }
  }
  // ---- stage wc tile transposed -> lds_wcT[c_local][p] (f32 -> bf16) ----
  {
    const float* src = wc + (size_t)b * (P_ * C_) + c0;
    for (int q = tid; q < 4096; q += 256) {
      const int p = q >> 4, x = (q & 15) * 4;
      f32x4 f = *(const f32x4*)(src + (size_t)p * C_ + x);
      cvt2_store(lds_wcT + (x    ) * WCT_STRIDE + p,
                 lds_wcT + (x + 1) * WCT_STRIDE + p, f[0], f[1]);
      cvt2_store(lds_wcT + (x + 2) * WCT_STRIDE + p,
                 lds_wcT + (x + 3) * WCT_STRIDE + p, f[2], f[3]);
    }
  }
  async_wait0();
  __syncthreads();

  // ---- GEMM1: one 16(c)x16(s) tile per wave; K=256 in 8 steps ----
  {
    const int mt = wv & 3;
    const int nt = wv >> 2;
    const unsigned short* arow = lds_wcT + (mt * 16 + ln) * WCT_STRIDE; // A = wc^T
    const unsigned short* brow = lds_we  + (nt * 16 + ln) * WE_STRIDE;  // B = we^T
    const int n = nt * 16 + ln;
    f32x8 acc = {0.f,0.f,0.f,0.f,0.f,0.f,0.f,0.f};
    for (int kk = 0; kk < 8; ++kk) {
      const int k0 = kk * 32;
      const unsigned short* ab = arow + k0 + 8 * half;
      const unsigned short* bb = brow + k0 + 8 * half;
      bf16x16 afr = frag_ld(ab, ab + 16);
      bf16x16 bfr = frag_ld(bb, bb + 16);
      acc = __builtin_amdgcn_wmma_f32_16x16x32_bf16(
          false, afr, false, bfr, (short)0, acc, false, false);
    }
    #pragma unroll
    for (int i = 0; i < 8; ++i)
      lds_lg[(mt * 16 + 8 * half + i) * 32 + n] = acc[i];
  }
  __syncthreads();

  // ---- softmax over s=32 per c-row; f32 attn -> global, bf16 -> LDS ----
  if (tid < CT_) {
    float v[32];
    float mx = -3.402823466e38f;
    #pragma unroll
    for (int s = 0; s < 32; ++s) { v[s] = lds_lg[tid * 32 + s]; mx = fmaxf(mx, v[s]); }
    float sum = 0.f;
    #pragma unroll
    for (int s = 0; s < 32; ++s) { v[s] = __expf(v[s] - mx); sum += v[s]; }
    const float inv = 1.0f / sum;
    float* gdst = attn_out + ((size_t)b * C_ + c0 + tid) * S_;
    unsigned int* arow = (unsigned int*)(lds_attn + tid * ATT_STRIDE);
    #pragma unroll
    for (int s = 0; s < 32; s += 2) {
      const float p0 = v[s] * inv, p1 = v[s + 1] * inv;
      gdst[s] = p0;
      gdst[s + 1] = p1;
      arow[s >> 1] = pack2_bf16(p0, p1);
    }
  }
  __syncthreads();

  // ---- GEMM2: out[p,c] = we^T(256x32) @ attn^T(32x64); K=32 -> 1 WMMA/tile ----
  {
    const int nt = wv & 3;
    const int n  = nt * 16 + ln;                       // c_local
    const unsigned short* bb = lds_attn + n * ATT_STRIDE + 8 * half;
    bf16x16 bfr = frag_ld(bb, bb + 16);                // wave-invariant B
    const int mtbase = (wv >> 2) * 8;
    for (int t = 0; t < 8; ++t) {
      const int mt = mtbase + t;
      const unsigned short* ab = lds_weT + (mt * 16 + ln) * WET_STRIDE + 8 * half;
      bf16x16 afr = frag_ld(ab, ab + 16);
      f32x8 acc = {0.f,0.f,0.f,0.f,0.f,0.f,0.f,0.f};
      acc = __builtin_amdgcn_wmma_f32_16x16x32_bf16(
          false, afr, false, bfr, (short)0, acc, false, false);
      #pragma unroll
      for (int i = 0; i < 8; ++i) {
        const int p = mt * 16 + 8 * half + i;
        out[((size_t)b * P_ + p) * C_ + c0 + n] = acc[i];
      }
    }
  }
}

// =====================================================================
extern "C" void kernel_launch(void* const* d_in, const int* in_sizes, int n_in,
                              void* d_out, int out_size, void* d_ws, size_t ws_size,
                              hipStream_t stream) {
  const float* wc        = (const float*)d_in[0];  // [128,16,16,1024]
  const float* word_emb  = (const float*)d_in[1];  // [128,32,256]
  const float* ck        = (const float*)d_in[2];  // [256,256]
  const float* bias      = (const float*)d_in[3];  // [256]

  float* out      = (float*)d_out;                       // [128,16,16,1024]
  float* attn_out = out + (size_t)B_ * P_ * C_;          // [128,1024,32]

  unsigned short* we_bf16  = (unsigned short*)d_ws;                       // 2 MB
  unsigned short* weT_bf16 = we_bf16 + (size_t)B_ * S_ * IDF_;            // 2 MB

  const size_t smem1 = (size_t)S_ * WE_STRIDE * 2 + (size_t)IDF_ * KT_STRIDE * 2;  // 37376
  const size_t smem2 = 84480;

  we_proj_kernel<<<B_, 256, smem1, stream>>>(word_emb, ck, bias, we_bf16, weT_bf16);
  fused_attn_kernel<<<B_ * (C_ / CT_), 256, smem2, stream>>>(
      wc, we_bf16, weT_bf16, out, attn_out);
}
